// BitNetAttention_65266323030286
// MI455X (gfx1250) — compile-verified, matrix operand order checked by reading, not run on hardware
//
#include <hip/hip_runtime.h>
#include <hip/hip_bf16.h>

typedef __bf16 bf16;
typedef __attribute__((ext_vector_type(16))) __bf16 v16bf;
typedef __attribute__((ext_vector_type(8)))  float  v8f;
typedef __attribute__((ext_vector_type(4)))  unsigned int u32x4;
typedef __attribute__((ext_vector_type(4)))  float  f32x4;
typedef __attribute__((ext_vector_type(4)))  __bf16 bf16x4;

// A/B fragment: 16 bf16 per lane as two 16B chunks (ISA 7.12.2 16-bit A layout)
union AB { v16bf v; u32x4 q[2]; };
// C/D fragment: 8 fp32 per lane
union F8 { v8f v; float f[8]; };

__device__ __forceinline__ bf16 f2bf(float f) {
  unsigned u = __builtin_bit_cast(unsigned, f);
  u += 0x7FFFu + ((u >> 16) & 1u);                    // round-to-nearest-even
  unsigned short h = (unsigned short)(u >> 16);
  return __builtin_bit_cast(bf16, h);
}

__device__ __forceinline__ v8f wmma_bf16(v16bf a, v16bf b, v8f c) {
  // (neg_a, A, neg_b, B, c_mod, C, reuse_a, reuse_b)
  return __builtin_amdgcn_wmma_f32_16x16x32_bf16(false, a, false, b, (short)0, c,
                                                 false, false);
}

// ---------------------------------------------------------------- fp32->bf16
__global__ __launch_bounds__(256) void cvt_f32_bf16(const float* __restrict__ src,
                                                    bf16* __restrict__ dst, int n4) {
  int i = blockIdx.x * 256 + threadIdx.x;
  const int stride = gridDim.x * 256;
  for (; i < n4; i += stride) {
    f32x4 v = ((const f32x4*)src)[i];
    bf16x4 o;
    o.x = f2bf(v.x); o.y = f2bf(v.y); o.z = f2bf(v.z); o.w = f2bf(v.w);
    ((bf16x4*)dst)[i] = o;
  }
}

// ------------------------------------------------- C[M,N] = A[M,K] * W[N,K]^T
// 256 threads = 8 waves (4 wave-rows x 2 wave-cols).
// Block tile 128x128, wave tile 32x64 (2x4 WMMA tiles -> 8 WMMA per K-step from
// 6 fragment loads). Double-buffered LDS: one barrier per K-step; next tile's
// global loads are issued before the WMMAs, its LDS stores after them.
__global__ __launch_bounds__(256) void gemm_bf16_nt(const bf16* __restrict__ A,
                                                    const bf16* __restrict__ W,
                                                    float* __restrict__ C,
                                                    int M, int N, int K) {
  __shared__ __attribute__((aligned(16))) bf16 sA[2][128][32];
  __shared__ __attribute__((aligned(16))) bf16 sB[2][128][32];
  const int wave = threadIdx.x >> 5;
  const int lane = threadIdx.x & 31;
  const int r = lane & 15, hi = lane >> 4;
  const int wr = wave >> 1, wc = wave & 1;
  const int m0 = blockIdx.y * 128, n0 = blockIdx.x * 128;

  // loader: 128x32 bf16 tile = 512 8-elem chunks; each thread owns 2 chunks
  const int lrow = threadIdx.x >> 2;                  // 0..63 (and +64)
  const int lcol = (threadIdx.x & 3) * 8;             // 0,8,16,24
  const bf16* aP0 = A + (size_t)(m0 + lrow) * K + lcol;
  const bf16* aP1 = A + (size_t)(m0 + lrow + 64) * K + lcol;
  const bf16* bP0 = W + (size_t)(n0 + lrow) * K + lcol;
  const bf16* bP1 = W + (size_t)(n0 + lrow + 64) * K + lcol;

  F8 acc[2][4];
#pragma unroll
  for (int ii = 0; ii < 2; ++ii)
#pragma unroll
    for (int j = 0; j < 4; ++j) acc[ii][j] = F8{};

  // prologue: stage first K tile into buffer 0
  *(u32x4*)&sA[0][lrow][lcol]      = *(const u32x4*)aP0;
  *(u32x4*)&sA[0][lrow + 64][lcol] = *(const u32x4*)aP1;
  *(u32x4*)&sB[0][lrow][lcol]      = *(const u32x4*)bP0;
  *(u32x4*)&sB[0][lrow + 64][lcol] = *(const u32x4*)bP1;
  __syncthreads();

  int p = 0;
  for (int k0 = 0; k0 < K; k0 += 32) {
    const bool more = (k0 + 32 < K);
    u32x4 la0{}, la1{}, lb0{}, lb1{};
    if (more) {                                       // issue next tile's loads early
      aP0 += 32; aP1 += 32; bP0 += 32; bP1 += 32;
      la0 = *(const u32x4*)aP0; la1 = *(const u32x4*)aP1;
      lb0 = *(const u32x4*)bP0; lb1 = *(const u32x4*)bP1;
      __builtin_prefetch(aP0 + 32, 0, 1);             // warm L2 one tile further out
      __builtin_prefetch(bP0 + 32, 0, 1);
    }
    AB a0, a1, b[4];
    a0.q[0] = *(const u32x4*)&sA[p][wr * 32 + r][hi * 8];
    a0.q[1] = *(const u32x4*)&sA[p][wr * 32 + r][16 + hi * 8];
    a1.q[0] = *(const u32x4*)&sA[p][wr * 32 + 16 + r][hi * 8];
    a1.q[1] = *(const u32x4*)&sA[p][wr * 32 + 16 + r][16 + hi * 8];
#pragma unroll
    for (int j = 0; j < 4; ++j) {
      b[j].q[0] = *(const u32x4*)&sB[p][wc * 64 + j * 16 + r][hi * 8];
      b[j].q[1] = *(const u32x4*)&sB[p][wc * 64 + j * 16 + r][16 + hi * 8];
    }
#pragma unroll
    for (int j = 0; j < 4; ++j) {
      acc[0][j].v = wmma_bf16(a0.v, b[j].v, acc[0][j].v);
      acc[1][j].v = wmma_bf16(a1.v, b[j].v, acc[1][j].v);
    }
    if (more) {                                       // stage next tile into other buffer
      const int q = p ^ 1;
      *(u32x4*)&sA[q][lrow][lcol]      = la0;
      *(u32x4*)&sA[q][lrow + 64][lcol] = la1;
      *(u32x4*)&sB[q][lrow][lcol]      = lb0;
      *(u32x4*)&sB[q][lrow + 64][lcol] = lb1;
    }
    __syncthreads();
    p ^= 1;
  }

#pragma unroll
  for (int ii = 0; ii < 2; ++ii)
#pragma unroll
    for (int j = 0; j < 4; ++j) {
      const int crow = m0 + wr * 32 + ii * 16 + hi * 8;
      const int ccol = n0 + wc * 64 + j * 16 + r;
#pragma unroll
      for (int i = 0; i < 8; ++i)
        C[(size_t)(crow + i) * N + ccol] = acc[ii][j].f[i];
    }
}

// ------------------------------------- RoPE + split + layout + bf16 conversion
// qkv [B*S,3072] fp32 -> Qr[b][h][s][d] (pre-scaled by HD^-0.5), Kr[b][hk][s][d],
// Vt[b][hv][d][s] (transposed so P*V B-fragments are contiguous), all bf16.
__global__ __launch_bounds__(256) void rope_split(const float* __restrict__ qkv,
                                                  const int* __restrict__ pos,
                                                  bf16* __restrict__ Qr,
                                                  bf16* __restrict__ Kr,
                                                  bf16* __restrict__ Vt) {
  const int token = blockIdx.x;                       // 0..2047
  const int b = token >> 10, s = token & 1023;
  const float p = (float)pos[token];
  const float* base = qkv + (size_t)token * 3072;
  for (int idx = threadIdx.x; idx < 1536; idx += 256) {
    const int head = idx >> 5;                        // 0..47
    const int i = idx & 31;                           // rotation pair index
    const float x1 = base[head * 64 + i];
    const float x2 = base[head * 64 + 32 + i];
    if (head < 40) {                                  // q or k: apply NeoX RoPE
      const float freq = __powf(500000.0f, -(float)(2 * i) * (1.0f / 64.0f));
      float sn, cs;
      __sincosf(p * freq, &sn, &cs);
      const float o1 = x1 * cs - x2 * sn;
      const float o2 = x2 * cs + x1 * sn;
      if (head < 32) {
        bf16* dst = Qr + (((size_t)(b * 32 + head)) * 1024 + s) * 64;
        dst[i]      = f2bf(o1 * 0.125f);              // fold in softmax scale
        dst[32 + i] = f2bf(o2 * 0.125f);
      } else {
        bf16* dst = Kr + (((size_t)(b * 8 + head - 32)) * 1024 + s) * 64;
        dst[i]      = f2bf(o1);
        dst[32 + i] = f2bf(o2);
      }
    } else {                                          // v: transpose store
      bf16* dst = Vt + ((size_t)(b * 8 + head - 40)) * 64 * 1024;
      dst[(size_t)i * 1024 + s]        = f2bf(x1);
      dst[(size_t)(32 + i) * 1024 + s] = f2bf(x2);
    }
  }
}

// ----------------------------------------------- causal GQA flash attention
// Grid: B*NH*(S/64) blocks of 128 threads; each wave owns one 16-row Q tile.
__global__ __launch_bounds__(128) void attn_fwd(const bf16* __restrict__ Qr,
                                                const bf16* __restrict__ Kr,
                                                const bf16* __restrict__ Vt,
                                                float* __restrict__ attnOut) {
  __shared__ __attribute__((aligned(16))) bf16 sP[4][16][32];   // per-wave P stage
  const int wave = threadIdx.x >> 5;
  const int lane = threadIdx.x & 31;
  const int r = lane & 15, hi = lane >> 4;
  const int bid = blockIdx.x;
  const int tg = bid & 15;                            // 16 groups of 64 q rows
  const int h  = (bid >> 4) & 31;                     // q head
  const int b  = bid >> 9;
  const int q0 = tg * 64 + wave * 16;
  const int hkv = h >> 2;                             // GQA group of 4

  const bf16* Qbase = Qr + (((size_t)(b * 32 + h)) * 1024 + q0) * 64;
  const bf16* Kbase = Kr + ((size_t)(b * 8 + hkv)) * 1024 * 64;
  const bf16* Vbase = Vt + ((size_t)(b * 8 + hkv)) * 64 * 1024;

  // Q A-fragments: dims 0..31 and 32..63 (K=32 each)
  AB aq0, aq1;
  {
    const bf16* qrow = Qbase + (size_t)r * 64;
    aq0.q[0] = *(const u32x4*)(qrow + hi * 8);
    aq0.q[1] = *(const u32x4*)(qrow + 16 + hi * 8);
    aq1.q[0] = *(const u32x4*)(qrow + 32 + hi * 8);
    aq1.q[1] = *(const u32x4*)(qrow + 48 + hi * 8);
  }

  F8 o0{}, o1{}, o2{}, o3{};                          // O tile 16x64
  float m_i[8], l_i[8];
#pragma unroll
  for (int i = 0; i < 8; ++i) { m_i[i] = -1e30f; l_i[i] = 0.f; }

  const int nblk = (q0 + 47) >> 5;                    // 32-key blocks, causal bound
  for (int blk = 0; blk < nblk; ++blk) {
    const int kk = blk * 32;
    // K B-fragments: column n = key kk+sub*16+n, contiguous head-dims
    AB bk00, bk01, bk10, bk11;
    {
      const bf16* k0p = Kbase + (size_t)(kk + r) * 64;
      bk00.q[0] = *(const u32x4*)(k0p + hi * 8);
      bk00.q[1] = *(const u32x4*)(k0p + 16 + hi * 8);
      bk01.q[0] = *(const u32x4*)(k0p + 32 + hi * 8);
      bk01.q[1] = *(const u32x4*)(k0p + 48 + hi * 8);
      const bf16* k1p = Kbase + (size_t)(kk + 16 + r) * 64;
      bk10.q[0] = *(const u32x4*)(k1p + hi * 8);
      bk10.q[1] = *(const u32x4*)(k1p + 16 + hi * 8);
      bk11.q[0] = *(const u32x4*)(k1p + 32 + hi * 8);
      bk11.q[1] = *(const u32x4*)(k1p + 48 + hi * 8);
    }
    F8 s0{}, s1{};                                    // scores 16x32 (two 16x16)
    s0.v = wmma_bf16(aq0.v, bk00.v, s0.v);
    s0.v = wmma_bf16(aq1.v, bk01.v, s0.v);
    s1.v = wmma_bf16(aq0.v, bk10.v, s1.v);
    s1.v = wmma_bf16(aq1.v, bk11.v, s1.v);

    // online softmax; C layout: row = q0+8*hi+i, col = kk(+16)+r
#pragma unroll
    for (int i = 0; i < 8; ++i) {
      const int qrow = q0 + hi * 8 + i;
      float v0 = (kk + r      <= qrow) ? s0.f[i] : -1e30f;
      float v1 = (kk + 16 + r <= qrow) ? s1.f[i] : -1e30f;
      float mx = fmaxf(v0, v1);
#pragma unroll
      for (int off = 1; off < 16; off <<= 1)          // xor<16 stays in half-wave
        mx = fmaxf(mx, __shfl_xor(mx, off, 32));
      const float mnew = fmaxf(m_i[i], mx);
      const float sf = __expf(m_i[i] - mnew);
      const float p0 = __expf(v0 - mnew);
      const float p1 = __expf(v1 - mnew);
      float rs = p0 + p1;
#pragma unroll
      for (int off = 1; off < 16; off <<= 1)
        rs += __shfl_xor(rs, off, 32);
      l_i[i] = l_i[i] * sf + rs;
      m_i[i] = mnew;
      o0.f[i] *= sf; o1.f[i] *= sf; o2.f[i] *= sf; o3.f[i] *= sf;
      sP[wave][hi * 8 + i][r]      = f2bf(p0);        // C-layout -> row-major LDS
      sP[wave][hi * 8 + i][16 + r] = f2bf(p1);
    }
    asm volatile("s_wait_dscnt 0x0" ::: "memory");    // same-wave DS RAW fence

    AB pf;                                            // reload P in A layout
    pf.q[0] = *(const u32x4*)&sP[wave][r][hi * 8];
    pf.q[1] = *(const u32x4*)&sP[wave][r][16 + hi * 8];

    // P(16x32) * V(32x64): V columns are contiguous thanks to transposed store
    AB vf;
    const bf16* vp = Vbase + (size_t)r * 1024 + kk;
    vf.q[0] = *(const u32x4*)(vp + hi * 8);
    vf.q[1] = *(const u32x4*)(vp + 16 + hi * 8);
    o0.v = wmma_bf16(pf.v, vf.v, o0.v);
    vp += 16 * 1024;
    vf.q[0] = *(const u32x4*)(vp + hi * 8);
    vf.q[1] = *(const u32x4*)(vp + 16 + hi * 8);
    o1.v = wmma_bf16(pf.v, vf.v, o1.v);
    vp += 16 * 1024;
    vf.q[0] = *(const u32x4*)(vp + hi * 8);
    vf.q[1] = *(const u32x4*)(vp + 16 + hi * 8);
    o2.v = wmma_bf16(pf.v, vf.v, o2.v);
    vp += 16 * 1024;
    vf.q[0] = *(const u32x4*)(vp + hi * 8);
    vf.q[1] = *(const u32x4*)(vp + 16 + hi * 8);
    o3.v = wmma_bf16(pf.v, vf.v, o3.v);
  }

#pragma unroll
  for (int i = 0; i < 8; ++i) {
    const float inv = 1.0f / l_i[i];
    float* row = attnOut + ((size_t)b * 1024 + q0 + hi * 8 + i) * 2048 + h * 64;
    row[r]      = o0.f[i] * inv;
    row[16 + r] = o1.f[i] * inv;
    row[32 + r] = o2.f[i] * inv;
    row[48 + r] = o3.f[i] * inv;
  }
}

// ------------------------------------------------------- RMSNorm -> bf16 rows
__global__ __launch_bounds__(256) void rmsnorm_bf16(const float* __restrict__ x,
                                                    const float* __restrict__ w,
                                                    bf16* __restrict__ out) {
  __shared__ float red[8];
  const int token = blockIdx.x;
  const float* row = x + (size_t)token * 2048;
  float ss = 0.f;
  for (int i = threadIdx.x; i < 2048; i += 256) { const float v = row[i]; ss += v * v; }
#pragma unroll
  for (int off = 16; off >= 1; off >>= 1) ss += __shfl_xor(ss, off, 32);
  if ((threadIdx.x & 31) == 0) red[threadIdx.x >> 5] = ss;
  __syncthreads();
  const float tot = red[0] + red[1] + red[2] + red[3] +
                    red[4] + red[5] + red[6] + red[7];
  const float inv = rsqrtf(tot * (1.0f / 2048.0f) + 1e-5f);
  bf16* orow = out + (size_t)token * 2048;
  for (int i = threadIdx.x; i < 2048; i += 256) orow[i] = f2bf(row[i] * inv * w[i]);
}

// ---------------------------------------------------------------------- launch
extern "C" void kernel_launch(void* const* d_in, const int* in_sizes, int n_in,
                              void* d_out, int out_size, void* d_ws, size_t ws_size,
                              hipStream_t stream) {
  const float* hidden    = (const float*)d_in[0];
  const int*   positions = (const int*)d_in[1];
  const float* w_qkv     = (const float*)d_in[2];
  const float* w_o       = (const float*)d_in[3];
  const float* norm_w    = (const float*)d_in[4];
  float* out = (float*)d_out;
  (void)in_sizes; (void)n_in; (void)out_size; (void)ws_size;

  constexpr int B = 2, S = 1024, H = 2048, NH = 32, NKV = 8, HD = 64;
  constexpr int M = B * S;                       // 2048 tokens
  constexpr int QKVN = (NH + 2 * NKV) * HD;      // 3072

  char* ws = (char*)d_ws;
  size_t off = 0;
  auto take = [&](size_t bytes) -> void* {
    void* p = ws + off;
    off = (off + bytes + 255) & ~(size_t)255;
    return p;
  };
  bf16*  hbf  = (bf16*)take((size_t)M * H * 2);
  bf16*  wqbf = (bf16*)take((size_t)QKVN * H * 2);
  bf16*  wobf = (bf16*)take((size_t)H * H * 2);
  float* qkv  = (float*)take((size_t)M * QKVN * 4);
  bf16*  Qr   = (bf16*)take((size_t)B * NH * S * HD * 2);
  bf16*  Kr   = (bf16*)take((size_t)B * NKV * S * HD * 2);
  bf16*  Vt   = (bf16*)take((size_t)B * NKV * HD * S * 2);
  float* attn = (float*)take((size_t)M * H * 4);
  bf16*  nrm  = (bf16*)take((size_t)M * H * 2);

  cvt_f32_bf16<<<512, 256, 0, stream>>>(hidden, hbf, M * H / 4);
  cvt_f32_bf16<<<512, 256, 0, stream>>>(w_qkv, wqbf, QKVN * H / 4);
  cvt_f32_bf16<<<512, 256, 0, stream>>>(w_o, wobf, H * H / 4);

  gemm_bf16_nt<<<dim3(QKVN / 128, M / 128), 256, 0, stream>>>(hbf, wqbf, qkv, M, QKVN, H);
  rope_split<<<M, 256, 0, stream>>>(qkv, positions, Qr, Kr, Vt);
  attn_fwd<<<B * NH * (S / 64), 128, 0, stream>>>(Qr, Kr, Vt, attn);
  rmsnorm_bf16<<<M, 256, 0, stream>>>(attn, norm_w, nrm);
  gemm_bf16_nt<<<dim3(H / 128, M / 128), 256, 0, stream>>>(nrm, wobf, out, M, H, H);
}
